// GraphModel_61194694033854
// MI455X (gfx1250) — matrix operand order repaired and checked by reading, not compiled, not run on hardware
//
#include <hip/hip_runtime.h>
#include <hip/hip_bf16.h>
#include <math.h>

// ---------------- types ----------------
typedef __attribute__((ext_vector_type(16))) __bf16 v16bf;
typedef __attribute__((ext_vector_type(8)))  float  v8f;
typedef __attribute__((ext_vector_type(4)))  unsigned int u32x4;
typedef __attribute__((ext_vector_type(4)))  float  f32x4;

union FragBF { v16bf v; u32x4 q[2]; };
union Pack16 { unsigned short h[16]; u32x4 q[2]; };

__device__ __forceinline__ unsigned short f32_to_bf16_rne(float f) {
    unsigned int u = __float_as_uint(f);
    u += 0x7FFFu + ((u >> 16) & 1u);
    return (unsigned short)(u >> 16);
}

__device__ __forceinline__ void atomicAddF32(float* p, float v) {
    __hip_atomic_fetch_add(p, v, __ATOMIC_RELAXED, __HIP_MEMORY_SCOPE_AGENT);
}

// ---------------- problem constants ----------------
#define NN      50000
#define NE      300000
#define IND     2000
#define DF      128
#define H1D     256
#define H2D     128
#define OUTD    32
#define KNUM    4
#define INDPAD  2016   // 63*32
#define BM      128    // block M tile

// ---------------- generic helpers ----------------
__global__ void fill_kernel(float* p, float v, int n) {
    int i = blockIdx.x * blockDim.x + threadIdx.x;
    if (i < n) p[i] = v;
}

__global__ void relu_kernel(float* p, int n) {
    int i = blockIdx.x * blockDim.x + threadIdx.x;
    if (i < n) p[i] = fmaxf(p[i], 0.0f);
}

// B [K x N] f32 row-major  ->  BT [N x Kpad] bf16 (zero padded in K)
__global__ void convert_bt_kernel(const float* __restrict__ B, unsigned short* __restrict__ BT,
                                  int K, int N, int Kpad) {
    int idx = blockIdx.x * blockDim.x + threadIdx.x;
    if (idx >= N * Kpad) return;
    int n = idx / Kpad, k = idx - n * Kpad;
    float f = (k < K) ? B[(size_t)k * N + n] : 0.0f;
    BT[idx] = f32_to_bf16_rne(f);
}

// ---------------- bf16 WMMA GEMM: C[MxN] = A[MxK] * B[KxN] ----------------
// A:  f32 row-major, converted to bf16 in LDS (BM x 32 tile per K-step)
// BT: bf16 [N][Kpad] (pre-transposed, K zero-padded to mult of 32)
// grid = (N/128, ceil(M/BM)); block = 256 threads (8 waves).
// Wave w owns rows [m0 + w*16, +16) and iterates all 8 N-subtiles:
//   8 x v_wmma_f32_16x16x32_bf16 per wave per K-step (A fragment reused 8x).
__global__ void gemm_bf16_kernel(const float* __restrict__ A,
                                 const unsigned short* __restrict__ BT,
                                 float* __restrict__ C,
                                 int M, int Kdim, int Kpad, int N) {
    __shared__ __align__(16) unsigned short aTile[BM * 32];   // 8 KB
    const int tid  = threadIdx.x;
    const int lane = tid & 31;
    const int wave = tid >> 5;
    const int m0   = blockIdx.y * BM;
    const int nb0  = blockIdx.x * 128;
    const int kgA  = (lane >> 4) * 8;    // A fragment K group base (halves)
    const int kgB  = (lane >> 4) * 8;    // B fragment K group base (halves)

    // staging assignment: thread t -> row t/2, 16 contiguous K at (t&1)*16
    const int  srow  = tid >> 1;
    const int  skk   = (tid & 1) * 16;
    const bool rowOK = (m0 + srow) < M;
    const float* aRow = A + (size_t)(m0 + srow) * Kdim + skk;
    unsigned short* sdst = &aTile[srow * 32 + skk];

    const unsigned short* arow = &aTile[(wave * 16 + (lane & 15)) * 32];
    const int colLane = lane & 15;

    v8f acc[8] = {};
    for (int k0 = 0; k0 < Kpad; k0 += 32) {
        // ---- stage A (vectorized; guard path only for K/M tails) ----
        Pack16 pk;
        if (rowOK && (k0 + skk + 16) <= Kdim) {
            const f32x4* p = (const f32x4*)(aRow + k0);
#pragma unroll
            for (int v = 0; v < 4; ++v) {
                f32x4 x = p[v];
#pragma unroll
                for (int j = 0; j < 4; ++j) pk.h[v * 4 + j] = f32_to_bf16_rne(x[j]);
            }
        } else {
#pragma unroll
            for (int j = 0; j < 16; ++j) {
                int k = k0 + skk + j;
                float fv = (rowOK && k < Kdim) ? aRow[k0 + j] : 0.0f;
                pk.h[j] = f32_to_bf16_rne(fv);
            }
        }
        *(u32x4*)(sdst)     = pk.q[0];
        *(u32x4*)(sdst + 8) = pk.q[1];
        __syncthreads();

        // ---- A fragment for this wave's 16-row strip ----
        FragBF a;
        a.q[0] = *(const u32x4*)(arow + kgA);        // K = kgA..kgA+7
        a.q[1] = *(const u32x4*)(arow + kgA + 16);   // K = kgA+16..kgA+23

        // ---- 8 N-subtiles, reuse A fragment ----
#pragma unroll
        for (int nt = 0; nt < 8; ++nt) {
            int col = nb0 + nt * 16 + colLane;
            const unsigned short* btcol = BT + (size_t)col * Kpad + k0 + kgB;
            FragBF b;
            b.q[0] = *(const u32x4*)(btcol);         // K = kgB..kgB+7
            b.q[1] = *(const u32x4*)(btcol + 16);    // K = kgB+16..kgB+23
            acc[nt] = __builtin_amdgcn_wmma_f32_16x16x32_bf16(
                false, a.v, false, b.v, (short)0, acc[nt], false, false);
        }
        __syncthreads();
    }

    // ---- store: lane -> col (lane&15), rows split by lane half ----
    const int rb = m0 + wave * 16 + (lane >> 4) * 8;
#pragma unroll
    for (int nt = 0; nt < 8; ++nt) {
        int col = nb0 + nt * 16 + colLane;
#pragma unroll
        for (int r = 0; r < 8; ++r) {
            int row = rb + r;
            if (row < M) C[(size_t)row * N + col] = acc[nt][r];
        }
    }
}

// ---------------- per-edge gate MLP (all K heads) + degree accumulation ------
__global__ void edge_gate_kernel(const float* __restrict__ h0,
                                 const int* __restrict__ src, const int* __restrict__ dst,
                                 const float* __restrict__ W1, const float* __restrict__ b1,
                                 const float* __restrict__ W2, const float* __restrict__ b2,
                                 float* __restrict__ Gk, float* __restrict__ deg) {
    __shared__ float sW1[KNUM * 2 * DF * 2];   // [k][f(256)][h(2)] = 2048 floats
    for (int i = threadIdx.x; i < KNUM * 2 * DF * 2; i += blockDim.x) sW1[i] = W1[i];
    __syncthreads();
    int e = blockIdx.x * blockDim.x + threadIdx.x;
    if (e >= NE) return;
    int s = src[e], d = dst[e];
    const f32x4* hs = (const f32x4*)(h0 + (size_t)s * DF);
    const f32x4* hd = (const f32x4*)(h0 + (size_t)d * DF);
    float acc[KNUM][2] = {};
    for (int q = 0; q < DF / 4; ++q) {
        f32x4 xs = hs[q], xd = hd[q];
#pragma unroll
        for (int j = 0; j < 4; ++j) {
            int f = q * 4 + j;
            float a = xs[j], bb = xd[j];
#pragma unroll
            for (int k = 0; k < KNUM; ++k) {
                const float* wk = &sW1[k * 512];
                acc[k][0] += a * wk[f * 2 + 0] + bb * wk[(DF + f) * 2 + 0];
                acc[k][1] += a * wk[f * 2 + 1] + bb * wk[(DF + f) * 2 + 1];
            }
        }
    }
#pragma unroll
    for (int k = 0; k < KNUM; ++k) {
        float hA = fmaxf(acc[k][0] + b1[k * 2 + 0], 0.0f);
        float hB = fmaxf(acc[k][1] + b1[k * 2 + 1], 0.0f);
        float z  = hA * W2[k * 2 + 0] + hB * W2[k * 2 + 1] + b2[k];
        float g  = 1.0f / (1.0f + expf(-z));
        Gk[(size_t)k * NE + e] = g;
        atomicAddF32(&deg[(size_t)k * NN + d], g);
    }
}

__global__ void dinv_kernel(const float* __restrict__ deg, float* __restrict__ dinv, int n) {
    int i = blockIdx.x * blockDim.x + threadIdx.x;
    if (i >= n) return;
    float d = deg[i];
    dinv[i] = (d > 0.0f) ? rsqrtf(d) : 0.0f;
}

__global__ void norm_kernel(const float* __restrict__ dinv, const float* __restrict__ Gk,
                            const int* __restrict__ src, const int* __restrict__ dst,
                            float* __restrict__ nrm, int total) {
    int idx = blockIdx.x * blockDim.x + threadIdx.x;
    if (idx >= total) return;
    int k = idx / NE, e = idx - k * NE;
    nrm[idx] = dinv[(size_t)k * NN + src[e]] * Gk[idx] * dinv[(size_t)k * NN + dst[e]];
}

// z[n][c] = bias[c] + dinv[n]^2 * t[n][c]   (self loop, weight 1)
__global__ void agg_init_kernel(const float* __restrict__ t, float* __restrict__ z,
                                const float* __restrict__ bias, const float* __restrict__ dinv,
                                int C, int total) {
    int idx = blockIdx.x * blockDim.x + threadIdx.x;
    if (idx >= total) return;
    int n = idx / C, c = idx - n * C;
    float di = dinv[n];
    z[idx] = bias[c] + di * di * t[idx];
}

// block per edge, thread per channel: z[dst][c] += norm[e] * t[src][c]
__global__ void agg_edge_kernel(const float* __restrict__ t, float* __restrict__ z,
                                const float* __restrict__ nrm,
                                const int* __restrict__ src, const int* __restrict__ dst, int C) {
    int e = blockIdx.x;
    float w = nrm[e];
    int s = src[e], d = dst[e];
    int c = threadIdx.x;
    atomicAddF32(&z[(size_t)d * C + c], w * t[(size_t)s * C + c]);
}

// mean over nodes of relu(z2): thread c accumulates, atomic to mean[c]
__global__ void mean_reduce_kernel(const float* __restrict__ z2, float* __restrict__ mean) {
    int c = threadIdx.x;  // H2D threads
    float s = 0.0f;
    for (int n = blockIdx.x; n < NN; n += gridDim.x)
        s += fmaxf(z2[(size_t)n * H2D + c], 0.0f);
    atomicAddF32(&mean[c], s);
}

__global__ void final_desc_kernel(const float* __restrict__ mean, const float* __restrict__ fcW,
                                  const float* __restrict__ fcb, float* __restrict__ out) {
    int o = threadIdx.x;  // OUTD threads
    float s = fcb[o];
    for (int c = 0; c < H2D; ++c)
        s += (mean[c] / (float)NN) * fcW[c * OUTD + o];
    out[o] = s;
}

// ---------------- host driver ----------------
extern "C" void kernel_launch(void* const* d_in, const int* in_sizes, int n_in,
                              void* d_out, int out_size, void* d_ws, size_t ws_size,
                              hipStream_t stream) {
    const float* X    = (const float*)d_in[0];
    const int*   ei   = (const int*)  d_in[1];
    const float* Wh   = (const float*)d_in[2];
    const float* W1   = (const float*)d_in[3];
    const float* b1   = (const float*)d_in[4];
    const float* W2   = (const float*)d_in[5];
    const float* b2   = (const float*)d_in[6];
    const float* g1W  = (const float*)d_in[7];
    const float* g1b  = (const float*)d_in[8];
    const float* g2W  = (const float*)d_in[9];
    const float* g2b  = (const float*)d_in[10];
    const float* fcW  = (const float*)d_in[11];
    const float* fcb  = (const float*)d_in[12];
    const int* src = ei;
    const int* dst = ei + NE;

    float* descOut = (float*)d_out;               // [K*32]
    float* GkOut   = (float*)d_out + KNUM * OUTD; // [K*E]

    // workspace layout
    size_t off = 0;
    auto wsAlloc = [&](size_t bytes) {
        void* p = (char*)d_ws + off;
        off = (off + bytes + 255) & ~(size_t)255;
        return p;
    };
    unsigned short* WhT  = (unsigned short*)wsAlloc((size_t)DF  * INDPAD * 2);
    unsigned short* g1WT = (unsigned short*)wsAlloc((size_t)H1D * DF     * 2);
    unsigned short* g2WT = (unsigned short*)wsAlloc((size_t)H2D * H1D    * 2);
    float* h0   = (float*)wsAlloc((size_t)NN * DF  * 4);
    float* t1   = (float*)wsAlloc((size_t)NN * H1D * 4);
    float* z1   = (float*)wsAlloc((size_t)NN * H1D * 4);
    float* t2   = (float*)wsAlloc((size_t)NN * H2D * 4);
    float* z2   = (float*)wsAlloc((size_t)NN * H2D * 4);
    float* deg  = (float*)wsAlloc((size_t)KNUM * NN * 4);
    float* dinv = (float*)wsAlloc((size_t)KNUM * NN * 4);
    float* nrm  = (float*)wsAlloc((size_t)KNUM * NE * 4);
    float* mean = (float*)wsAlloc(H2D * 4);

    const int T = 256;
    auto nb = [](int n, int t) { return (n + t - 1) / t; };
    const int MB = (NN + BM - 1) / BM;   // 391 row-blocks

    // weight transposes to bf16 [N][Kpad]
    convert_bt_kernel<<<nb(DF * INDPAD, T), T, 0, stream>>>(Wh,  WhT,  IND, DF,  INDPAD);
    convert_bt_kernel<<<nb(H1D * DF, T),   T, 0, stream>>>(g1W, g1WT, DF,  H1D, DF);
    convert_bt_kernel<<<nb(H2D * H1D, T),  T, 0, stream>>>(g2W, g2WT, H1D, H2D, H1D);

    // h0 = X @ Wh   (M=50000, K=2000->2016, N=128)
    gemm_bf16_kernel<<<dim3(1, MB), T, 0, stream>>>(X, WhT, h0, NN, IND, INDPAD, DF);

    // degrees start at 1 (self loops)
    fill_kernel<<<nb(KNUM * NN, T), T, 0, stream>>>(deg, 1.0f, KNUM * NN);

    // per-edge gate MLP -> Gk (all k), accumulate deg
    edge_gate_kernel<<<nb(NE, T), T, 0, stream>>>(h0, src, dst, W1, b1, W2, b2, GkOut, deg);

    dinv_kernel<<<nb(KNUM * NN, T), T, 0, stream>>>(deg, dinv, KNUM * NN);
    norm_kernel<<<nb(KNUM * NE, T), T, 0, stream>>>(dinv, GkOut, src, dst, nrm, KNUM * NE);

    // t1 = h0 @ g1_W  (shared across k)
    gemm_bf16_kernel<<<dim3(H1D / 128, MB), T, 0, stream>>>(h0, g1WT, t1, NN, DF, DF, H1D);

    for (int k = 0; k < KNUM; ++k) {
        const float* dk = dinv + (size_t)k * NN;
        const float* nk = nrm  + (size_t)k * NE;
        // layer 1 aggregate
        agg_init_kernel<<<nb(NN * H1D, T), T, 0, stream>>>(t1, z1, g1b, dk, H1D, NN * H1D);
        agg_edge_kernel<<<NE, H1D, 0, stream>>>(t1, z1, nk, src, dst, H1D);
        relu_kernel<<<nb(NN * H1D, T), T, 0, stream>>>(z1, NN * H1D);
        // t2 = z1 @ g2_W
        gemm_bf16_kernel<<<dim3(H2D / 128, MB), T, 0, stream>>>(z1, g2WT, t2, NN, H1D, H1D, H2D);
        // layer 2 aggregate
        agg_init_kernel<<<nb(NN * H2D, T), T, 0, stream>>>(t2, z2, g2b, dk, H2D, NN * H2D);
        agg_edge_kernel<<<NE, H2D, 0, stream>>>(t2, z2, nk, src, dst, H2D);
        // mean(relu(z2)) -> descriptor
        fill_kernel<<<1, H2D, 0, stream>>>(mean, 0.0f, H2D);
        mean_reduce_kernel<<<256, H2D, 0, stream>>>(z2, mean);
        final_desc_kernel<<<1, OUTD, 0, stream>>>(mean, fcW, fcb, descOut + k * OUTD);
    }
}
